// GAT_65171833749734
// MI455X (gfx1250) — matrix operand order repaired and checked by reading, not compiled, not run on hardware
//
#include <hip/hip_runtime.h>

typedef __attribute__((ext_vector_type(16))) _Float16 v16h;
typedef __attribute__((ext_vector_type(8)))  float    v8f;

#define GFEAT 16
#define LDB   136   // padded LDS row stride (halves): 272B -> 4-bank skew, 16B aligned

// ---------------- utility ----------------
__global__ void k_f32_to_f16(const float* __restrict__ s, _Float16* __restrict__ d, int n) {
  for (int i = blockIdx.x * blockDim.x + threadIdx.x; i < n; i += gridDim.x * blockDim.x)
    d[i] = (_Float16)s[i];
}

__global__ void k_fill_u32(unsigned* __restrict__ p, unsigned v, int n) {
  for (int i = blockIdx.x * blockDim.x + threadIdx.x; i < n; i += gridDim.x * blockDim.x)
    p[i] = v;
}

// ------- GEMM: C[Mx128] = A[Mx128] * B[128x128]; f16 in, f32 out (WMMA) -------
__global__ __launch_bounds__(256) void k_gemm_wmma(
    const _Float16* __restrict__ A, const _Float16* __restrict__ B,
    float* __restrict__ C, int M) {
  __shared__ _Float16 Bs[128 * LDB];
  const int tid = threadIdx.x;
  // cooperative stage of B (128x128 f16 = 32KB) into LDS, padded rows
  for (int i = tid; i < 128 * 16; i += 256) {
    int row = i >> 4, c = (i & 15) << 3;
    *reinterpret_cast<int4*>(&Bs[row * LDB + c]) =
        *reinterpret_cast<const int4*>(B + row * 128 + c);
  }
  __syncthreads();

  const int wave = tid >> 5;
  const int lane = tid & 31;
  const int rowTile = blockIdx.x * 8 + wave;
  if (rowTile * 16 >= M) return;            // wave-uniform exit, EXEC stays full
  const int row0 = rowTile * 16;

  // ---- A fragments: 16x32 f16 per K-chunk. Lane l (l&15 = row M), lanes>=16
  // carry K offset +8; vgpr v holds K pair (v>>2)*16 + (v&3)*2 (+hk).
  const int r  = lane & 15;
  const int hk = (lane >> 4) * 8;
  int arow = row0 + r; if (arow >= M) arow = M - 1;
  const _Float16* Ap = A + (size_t)arow * 128;

  v16h af[4];
#pragma unroll
  for (int kc = 0; kc < 4; ++kc) {
#pragma unroll
    for (int v = 0; v < 8; ++v) {
      int K = kc * 32 + ((v >> 2) * 16) + hk + ((v & 3) * 2);
      af[kc][2 * v]     = Ap[K];
      af[kc][2 * v + 1] = Ap[K + 1];
    }
  }

  // C/D layout: vgpr i -> M = (lane>>4)*8 + i, N = lane&15
  const int mrow = (lane >> 4) * 8;
  const int ncol = lane & 15;
  const bool full = (row0 + 16 <= M);       // wave-uniform: tile fully in-bounds

#pragma unroll
  for (int nt = 0; nt < 8; ++nt) {
    const int n0 = nt * 16;
    v8f acc = {};
#pragma unroll
    for (int kc = 0; kc < 4; ++kc) {
      // B fragment 32x16: lane l holds column N = l&15, element j = K (j>>3)*16 + hk + (j&7)
      const _Float16* Bp = &Bs[(size_t)(kc * 32 + hk) * LDB + n0 + (lane & 15)];
      v16h bf;
#pragma unroll
      for (int j = 0; j < 16; ++j) bf[j] = Bp[(size_t)((j >> 3) * 16 + (j & 7)) * LDB];
      acc = __builtin_amdgcn_wmma_f32_16x16x32_f16(false, af[kc], false, bf,
                                                   (short)0, acc, false, false);
    }
    float* Cp = C + (size_t)(row0 + mrow) * 128 + n0 + ncol;
    if (full) {
      // fast path: no per-row guards -> straight-line global_store_b32 x8
#pragma unroll
      for (int i = 0; i < 8; ++i) Cp[(size_t)i * 128] = acc[i];
    } else {
#pragma unroll
      for (int i = 0; i < 8; ++i)
        if (row0 + mrow + i < M) Cp[(size_t)i * 128] = acc[i];
    }
  }
}

// ---------------- attention coefficients: es = h.a_src, ed = h.a_dst ----------
__global__ void k_attn(const float* __restrict__ h, const float* __restrict__ a_s,
                       const float* __restrict__ a_d, float* __restrict__ es,
                       float* __restrict__ ed, int N) {
  int t = blockIdx.x * blockDim.x + threadIdx.x;
  int node = t >> 5;
  if (node >= N) return;
  int lane = threadIdx.x & 31;
  float4 v  = reinterpret_cast<const float4*>(h + (size_t)node * 128)[lane];
  float4 s4 = reinterpret_cast<const float4*>(a_s)[lane];
  float4 d4 = reinterpret_cast<const float4*>(a_d)[lane];
  float s = v.x * s4.x + v.y * s4.y + v.z * s4.z + v.w * s4.w;
  float d = v.x * d4.x + v.y * d4.y + v.z * d4.z + v.w * d4.w;
#pragma unroll
  for (int o = 16; o > 0; o >>= 1) {
    s += __shfl_xor(s, o, 32);
    d += __shfl_xor(d, o, 32);
  }
  if (lane == 0) { es[node] = s; ed[node] = d; }
}

// ---------------- edge helpers ----------------
__device__ __forceinline__ void edge_sd(const int* __restrict__ ei, int i, int E,
                                        int& s, int& d) {
  if (i < E) { s = ei[i]; d = ei[E + i]; } else { s = d = i - E; }  // self loops
}
__device__ __forceinline__ float leaky(float e) { return e > 0.f ? e : 0.2f * e; }
__device__ __forceinline__ unsigned ord_enc(float f) {
  unsigned u = __float_as_uint(f);
  return (u & 0x80000000u) ? ~u : (u | 0x80000000u);
}
__device__ __forceinline__ float ord_dec(unsigned u) {
  return (u & 0x80000000u) ? __uint_as_float(u & 0x7FFFFFFFu) : __uint_as_float(~u);
}

__global__ void k_edge_max(const int* __restrict__ ei, const float* __restrict__ es,
                           const float* __restrict__ ed, unsigned* __restrict__ m,
                           int E, int N) {
  int i = blockIdx.x * blockDim.x + threadIdx.x;
  if (i >= E + N) return;
  int s, d; edge_sd(ei, i, E, s, d);
  atomicMax(&m[d], ord_enc(leaky(es[s] + ed[d])));
}

__global__ void k_edge_expsum(const int* __restrict__ ei, const float* __restrict__ es,
                              const float* __restrict__ ed, const unsigned* __restrict__ m,
                              float* __restrict__ ssum, float* __restrict__ exb,
                              int E, int N) {
  int i = blockIdx.x * blockDim.x + threadIdx.x;
  if (i >= E + N) return;
  int s, d; edge_sd(ei, i, E, s, d);
  float e = leaky(es[s] + ed[d]);
  float ex = __expf(e - ord_dec(m[d]));
  exb[i] = ex;
  atomicAdd(&ssum[d], ex);
}

__global__ void k_recip(const float* __restrict__ s, float* __restrict__ r, int N) {
  int i = blockIdx.x * blockDim.x + threadIdx.x;
  if (i < N) r[i] = 1.0f / s[i];   // self-loop guarantees s > 0
}

// one wave per edge; lane owns 4 features (coalesced 512B gather of h[src])
__global__ void k_edge_aggr(const int* __restrict__ ei, const float* __restrict__ h,
                            const float* __restrict__ exb, const float* __restrict__ sinv,
                            float* __restrict__ O, int E, int N) {
  int t = blockIdx.x * blockDim.x + threadIdx.x;
  int edge = t >> 5;
  if (edge >= E + N) return;
  int lane = threadIdx.x & 31;
  int s, d; edge_sd(ei, edge, E, s, d);
  float alpha = exb[edge] * sinv[d];
  float4 v = reinterpret_cast<const float4*>(h + (size_t)s * 128)[lane];
  float* Od = O + (size_t)d * 128 + lane * 4;
  atomicAdd(Od + 0, alpha * v.x);
  atomicAdd(Od + 1, alpha * v.y);
  atomicAdd(Od + 2, alpha * v.z);
  atomicAdd(Od + 3, alpha * v.w);
}

// relu(O + b) -> f32 (pooling input) and f16 (next layer's GEMM A)
__global__ void k_finalize(const float* __restrict__ O, const float* __restrict__ b,
                           float* __restrict__ hf32, _Float16* __restrict__ hf16, int n) {
  for (int i = blockIdx.x * blockDim.x + threadIdx.x; i < n; i += gridDim.x * blockDim.x) {
    float v = O[i] + b[i & 127];
    v = v > 0.f ? v : 0.f;
    hf32[i] = v;
    hf16[i] = (_Float16)v;
  }
}

// ---------------- pooling + head ----------------
__global__ void k_count(const int* __restrict__ batch, float* __restrict__ cnt, int N) {
  int i = blockIdx.x * blockDim.x + threadIdx.x;
  if (i < N) atomicAdd(&cnt[batch[i]], 1.0f);
}

__global__ void k_pool_sum(const float* __restrict__ h, const int* __restrict__ batch,
                           float* __restrict__ sums, int N) {
  int t = blockIdx.x * blockDim.x + threadIdx.x;
  int node = t >> 5;
  if (node >= N) return;
  int lane = threadIdx.x & 31;
  int b = batch[node];
  float4 v = reinterpret_cast<const float4*>(h + (size_t)node * 128)[lane];
  float* sp = sums + (size_t)b * 128 + lane * 4;
  atomicAdd(sp + 0, v.x);
  atomicAdd(sp + 1, v.y);
  atomicAdd(sp + 2, v.z);
  atomicAdd(sp + 3, v.w);
}

__global__ void k_head(const float* __restrict__ sums, const float* __restrict__ cnt,
                       const float* __restrict__ gfeat, const float* __restrict__ linW,
                       const float* __restrict__ linb, float* __restrict__ out,
                       int G, int C) {
  int t = blockIdx.x * blockDim.x + threadIdx.x;
  if (t >= G * C) return;
  int g = t / C, c = t % C;
  float rc = 1.0f / fmaxf(cnt[g], 1.0f);
  float acc = linb[c];
  for (int k = 0; k < 128; ++k) acc += sums[(size_t)g * 128 + k] * rc * linW[k * C + c];
  for (int j = 0; j < GFEAT; ++j) acc += gfeat[g * GFEAT + j] * linW[(128 + j) * C + c];
  out[t] = acc;
}

// ---------------- launcher ----------------
static inline int grd(int n) { return (n + 255) / 256; }

extern "C" void kernel_launch(void* const* d_in, const int* in_sizes, int n_in,
                              void* d_out, int out_size, void* d_ws, size_t ws_size,
                              hipStream_t stream) {
  (void)n_in; (void)out_size; (void)ws_size;
  const float* x     = (const float*)d_in[0];
  const int*   ei    = (const int*)d_in[1];
  const int*   batch = (const int*)d_in[2];
  const float* gfeat = (const float*)d_in[3];
  const float* Wm[2]  = {(const float*)d_in[4], (const float*)d_in[8]};
  const float* as_[2] = {(const float*)d_in[5], (const float*)d_in[9]};
  const float* ad_[2] = {(const float*)d_in[6], (const float*)d_in[10]};
  const float* b_[2]  = {(const float*)d_in[7], (const float*)d_in[11]};
  const float* linW  = (const float*)d_in[12];
  const float* linb  = (const float*)d_in[13];
  float* out = (float*)d_out;

  const int N  = in_sizes[2];
  const int E  = in_sizes[1] / 2;
  const int ET = E + N;
  const int G  = in_sizes[3] / GFEAT;
  const int C  = in_sizes[13];
  const int NH = N * 128;

  char* w = (char*)d_ws;
  size_t off = 0;
  auto alloc = [&](size_t bytes) -> void* {
    void* p = w + off;
    off = (off + bytes + 255) & ~(size_t)255;
    return p;
  };
  _Float16* Af16 = (_Float16*)alloc((size_t)NH * 2);
  _Float16* Wf16 = (_Float16*)alloc((size_t)128 * 128 * 2);
  float*    h    = (float*)alloc((size_t)NH * 4);
  float*    O    = (float*)alloc((size_t)NH * 4);
  float*    es   = (float*)alloc((size_t)N * 4);
  float*    ed   = (float*)alloc((size_t)N * 4);
  unsigned* m    = (unsigned*)alloc((size_t)N * 4);
  float*    ssum = (float*)alloc((size_t)N * 4);
  float*    sinv = (float*)alloc((size_t)N * 4);
  float*    exb  = (float*)alloc((size_t)ET * 4);
  float*    sums = (float*)alloc((size_t)G * 128 * 4);
  float*    cnt  = (float*)alloc((size_t)G * 4);

  const int TB = 256;
  const int rowTiles = (N + 15) / 16;

  k_f32_to_f16<<<grd(NH), TB, 0, stream>>>(x, Af16, NH);

  for (int L = 0; L < 2; ++L) {
    k_f32_to_f16<<<grd(128 * 128), TB, 0, stream>>>(Wm[L], Wf16, 128 * 128);
    k_gemm_wmma<<<(rowTiles + 7) / 8, 256, 0, stream>>>(Af16, Wf16, h, N);
    k_attn<<<grd(N * 32), TB, 0, stream>>>(h, as_[L], ad_[L], es, ed, N);
    k_fill_u32<<<grd(N), TB, 0, stream>>>(m, 0u, N);
    k_fill_u32<<<grd(N), TB, 0, stream>>>((unsigned*)ssum, 0u, N);
    k_fill_u32<<<grd(NH), TB, 0, stream>>>((unsigned*)O, 0u, NH);
    k_edge_max<<<grd(ET), TB, 0, stream>>>(ei, es, ed, m, E, N);
    k_edge_expsum<<<grd(ET), TB, 0, stream>>>(ei, es, ed, m, ssum, exb, E, N);
    k_recip<<<grd(N), TB, 0, stream>>>(ssum, sinv, N);
    k_edge_aggr<<<grd(ET * 32), TB, 0, stream>>>(ei, h, exb, sinv, O, E, N);
    k_finalize<<<grd(NH), TB, 0, stream>>>(O, b_[L], h, Af16, NH);
  }

  k_fill_u32<<<grd(G * 128), TB, 0, stream>>>((unsigned*)sums, 0u, G * 128);
  k_fill_u32<<<grd(G), TB, 0, stream>>>((unsigned*)cnt, 0u, G);
  k_count<<<grd(N), TB, 0, stream>>>(batch, cnt, N);
  k_pool_sum<<<grd(N * 32), TB, 0, stream>>>(h, batch, sums, N);
  k_head<<<grd(G * C), TB, 0, stream>>>(sums, cnt, gfeat, linW, linb, out, G, C);
}
